// ExRestSelfAtten_42571715838135
// MI455X (gfx1250) — matrix operand order repaired and checked by reading, not compiled, not run on hardware
//
#include <hip/hip_runtime.h>
#include <hip/hip_bf16.h>

// ---------------------------------------------------------------------------
// ExRestSelfAtten on MI455X (gfx1250, wave32, WMMA f32_16x16x32_f16)
//
// Algebra: keys/vals are shifted copies of (h@Wk)/(h@Wv) -> 4 GEMMs total.
// Bandwidth bound: x(41MB) + QKV(88MB f16) dominate; h never touches global
// memory at all -- the hidden GEMM and the QKV GEMM are fused through a 4KB
// LDS tile (h tile produced by phase-1 WMMAs is the A operand of phase 2).
// All fragment loads are unconditional contiguous vectors (b128).
// ---------------------------------------------------------------------------

typedef _Float16 v16h __attribute__((ext_vector_type(16)));
typedef _Float16 v8h  __attribute__((ext_vector_type(8)));
typedef _Float16 v4h  __attribute__((ext_vector_type(4)));
typedef float    v8f  __attribute__((ext_vector_type(8)));

#define BATCH 1024
#define SEQ   100
#define SPAD  112      // 7 tiles of 16
#define INF_  100      // IN feature dim (padded to 128 in Xh)
#define HDIM  128
#define MIDD  32
#define OUTD  2
#define NMTILE 7       // SPAD/16
#define SQRT_H 11.313708498984761f
#define LDS_STRIDE 136 // 128 + 8 halves: row-major A-frag reads hit 64 banks

#define REGH ((size_t)BATCH * SPAD * HDIM)   // halves per (B,112,128) buffer
#define WPACK 16384                          // halves per packed weight

__device__ __forceinline__ v8f wmma_f16(v16h a, v16h b, v8f c) {
  return __builtin_amdgcn_wmma_f32_16x16x32_f16(
      false, a, false, b, (short)0, c, false, false);
}

// A fragment (16x32 f16) from row-major f16 (global or LDS), lda elements.
// lane&15 = row; lane>>4 selects K sub-bank (0-7/16-23 vs 8-15/24-31).
template <typename PtrT>
__device__ __forceinline__ v16h load_a_f16(PtrT A, int lda, int k0, int lane) {
  int row = lane & 15;
  int kb  = (lane >> 4) << 3;
  v8h lo = *(const v8h*)(A + row * lda + k0 + kb);
  v8h hi = *(const v8h*)(A + row * lda + k0 + kb + 16);
  v16h a;
#pragma unroll
  for (int i = 0; i < 8; ++i) { a[i] = lo[i]; a[8 + i] = hi[i]; }
  return a;
}

// B fragment from pre-packed layout: one contiguous 32-byte load per lane.
__device__ __forceinline__ v16h load_b_packed(const _Float16* __restrict__ Wp,
                                              int kt, int nt, int lane) {
  return *(const v16h*)(Wp + (((size_t)(kt * 8 + nt) * 32) + lane) * 16);
}

// ---------------------------------------------------------------------------
// Prep A: x (B,100,100 f32) -> Xh (B,112,128 f16), zero padded rows/cols.
// ---------------------------------------------------------------------------
__global__ __launch_bounds__(256) void k_convert_x(
    const float* __restrict__ x, _Float16* __restrict__ Xh) {
  size_t idx = (size_t)blockIdx.x * 256 + threadIdx.x;   // < B*112*128
  int h = (int)(idx & (HDIM - 1));
  int s = (int)((idx >> 7) % SPAD);
  size_t b = idx / ((size_t)SPAD * HDIM);
  float v = 0.0f;
  if (s < SEQ && h < INF_) v = x[(b * SEQ + s) * INF_ + h];
  Xh[idx] = (_Float16)v;
}

// ---------------------------------------------------------------------------
// Prep B: pe (100,128 f32) -> pe_pad (112,128 f32), zero padded rows.
// ---------------------------------------------------------------------------
__global__ __launch_bounds__(256) void k_pad_pe(
    const float* __restrict__ pe, float* __restrict__ pe_pad) {
  int idx = blockIdx.x * 256 + threadIdx.x;   // < 112*128
  int s = idx >> 7;
  pe_pad[idx] = (s < SEQ) ? pe[idx] : 0.0f;
}

// ---------------------------------------------------------------------------
// Prep C: pack {W1,Wq,Wk,Wv} (f32 [K x 128]) -> WMMA B-fragment layout f16.
// K zero-padded to 128 (only W1 has K=100). 4096 threads, runs once.
// ---------------------------------------------------------------------------
__global__ __launch_bounds__(256) void k_pack_w(
    const float* __restrict__ W1, const float* __restrict__ Wq,
    const float* __restrict__ Wk, const float* __restrict__ Wv,
    _Float16* __restrict__ Wp) {
  int tid  = blockIdx.x * 256 + threadIdx.x;  // < 4096
  int mi   = tid >> 10;
  int rem  = tid & 1023;
  int kt   = rem >> 8;
  int nt   = (rem >> 5) & 7;
  int lane = rem & 31;

  const float* src = (mi == 0) ? W1 : (mi == 1) ? Wq : (mi == 2) ? Wk : Wv;
  int klim = (mi == 0) ? INF_ : HDIM;
  _Float16* dst = Wp + (size_t)mi * WPACK;

  int col = nt * 16 + (lane & 15);
  int kh  = (lane >> 4) << 3;
  v16h frag;
#pragma unroll
  for (int i = 0; i < 16; ++i) {
    int k = kt * 32 + kh + (i & 7) + ((i >> 3) << 4);
    frag[i] = (k < klim) ? (_Float16)src[k * HDIM + col] : (_Float16)0.0f;
  }
  *(v16h*)(dst + (((size_t)(kt * 8 + nt) * 32) + lane) * 16) = frag;
}

// ---------------------------------------------------------------------------
// Fused K1+K2: per block (b, 16-row M tile), 256 threads = 8 waves.
//  Phase 1: h = relu(Xh@W1 + b1) + pe  -> f16 tile in LDS (never to global)
//  Phase 2: Q/K/V = hT @ {Wq,Wk,Wv}    -> A frags from LDS (ds_load_b128)
// Each wave owns one 16-col N-tile; one A fragment feeds 3 WMMA chains.
// ---------------------------------------------------------------------------
__global__ __launch_bounds__(256) void k_hidden_qkv(
    const _Float16* __restrict__ Xh, const _Float16* __restrict__ Wp,
    const float* __restrict__ b1, const float* __restrict__ pe_pad,
    _Float16* __restrict__ Q, _Float16* __restrict__ K, _Float16* __restrict__ V) {
  __shared__ _Float16 hT[16 * LDS_STRIDE];   // 4.25 KB, bank-conflict-free

  int blk  = blockIdx.x;
  int b    = blk / NMTILE;
  int mt   = blk % NMTILE;
  int wave = threadIdx.x >> 5;
  int lane = threadIdx.x & 31;
  int col  = wave * 16 + (lane & 15);
  int rsub = (lane >> 4) * 8;
  int row_base = mt * 16;

  // ---- Phase 1: hidden GEMM ----
  const _Float16* A = Xh + ((size_t)b * SPAD + row_base) * HDIM;
  v8f acc = {};
#pragma unroll
  for (int kt = 0; kt < 4; ++kt) {
    v16h a  = load_a_f16(A, HDIM, kt * 32, lane);
    v16h bb = load_b_packed(Wp, kt, wave, lane);
    acc = wmma_f16(a, bb, acc);
  }
  float bias = b1[col];
#pragma unroll
  for (int i = 0; i < 8; ++i) {
    int s = row_base + rsub + i;
    float v = fmaxf(acc[i] + bias, 0.0f) + pe_pad[s * HDIM + col];
    v = (s < SEQ) ? v : 0.0f;                 // cndmask, no branch
    hT[(rsub + i) * LDS_STRIDE + col] = (_Float16)v;
  }
  __syncthreads();

  // ---- Phase 2: QKV GEMMs with A from LDS ----
  v8f aq = {}, ak = {}, av = {};
#pragma unroll
  for (int kt = 0; kt < 4; ++kt) {
    v16h a   = load_a_f16((const _Float16*)hT, LDS_STRIDE, kt * 32, lane);
    v16h bq  = load_b_packed(Wp + WPACK,     kt, wave, lane);
    v16h bk  = load_b_packed(Wp + 2 * WPACK, kt, wave, lane);
    v16h bv_ = load_b_packed(Wp + 3 * WPACK, kt, wave, lane);
    aq = wmma_f16(a, bq, aq);
    ak = wmma_f16(a, bk, ak);
    av = wmma_f16(a, bv_, av);
  }

  size_t base = ((size_t)b * SPAD + row_base) * HDIM;
#pragma unroll
  for (int i = 0; i < 8; ++i) {
    size_t idx = base + (size_t)(rsub + i) * HDIM + col;
    Q[idx] = (_Float16)aq[i];
    K[idx] = (_Float16)ak[i];
    V[idx] = (_Float16)av[i];
  }
}

// ---------------------------------------------------------------------------
// K3: windowed attention (5 neighbors) + softmax + context + 128->32->2 MLP.
// One wave per sequence row; s is wave-uniform so all window bounds checks
// are scalar branches. K/V rows read as v4h. Per-wave LDS slices (no block
// barrier -> safe early wave exit).
// Output layout: [B*S*2 floats output][B*S*5 floats softmax weights].
// ---------------------------------------------------------------------------
__global__ __launch_bounds__(256) void k_attn(
    const _Float16* __restrict__ Q, const _Float16* __restrict__ K,
    const _Float16* __restrict__ V,
    const float* __restrict__ bv, const float* __restrict__ W2,
    const float* __restrict__ b2, const float* __restrict__ W3,
    const float* __restrict__ b3, float* __restrict__ out) {
  __shared__ float lds_ctx[8][HDIM];
  __shared__ float lds_mid[8][MIDD];

  int blk  = blockIdx.x;
  int b    = blk / 13;
  int st   = blk % 13;
  int wave = threadIdx.x >> 5;
  int lane = threadIdx.x & 31;
  int s    = st * 8 + wave;
  if (s >= SEQ) return;

  size_t brow = (size_t)b * SPAD;
  v4h qv = *((const v4h*)(Q + (brow + s) * HDIM) + lane);
  float q[4];
#pragma unroll
  for (int i = 0; i < 4; ++i) q[i] = (float)qv[i];

  float sc[5];
#pragma unroll
  for (int m = 0; m < 5; ++m) {
    int j = s + 2 - m;                     // wave-uniform
    float p = 0.0f;
    if (j >= 0 && j < SEQ) {               // scalar branch
      v4h kv = *((const v4h*)(K + (brow + j) * HDIM) + lane);
#pragma unroll
      for (int i = 0; i < 4; ++i) p += q[i] * (float)kv[i];
    }
#pragma unroll
    for (int off = 16; off >= 1; off >>= 1) p += __shfl_xor(p, off, 32);
    sc[m] = p * (1.0f / SQRT_H);
  }

  float mx = sc[0];
#pragma unroll
  for (int m = 1; m < 5; ++m) mx = fmaxf(mx, sc[m]);
  float wgt[5], sum = 0.0f;
#pragma unroll
  for (int m = 0; m < 5; ++m) { wgt[m] = __expf(sc[m] - mx); sum += wgt[m]; }
  float inv = 1.0f / sum;
#pragma unroll
  for (int m = 0; m < 5; ++m) wgt[m] *= inv;

  if (lane < 5)
    out[(size_t)BATCH * SEQ * OUTD + ((size_t)b * SEQ + s) * 5 + lane] = wgt[lane];

  // context = sum_m wgt[m]*V[j] + bv   (sum(wgt)==1 folds bv exactly)
  float ctx[4];
#pragma unroll
  for (int i = 0; i < 4; ++i) ctx[i] = bv[lane * 4 + i];
#pragma unroll
  for (int m = 0; m < 5; ++m) {
    int j = s + 2 - m;
    if (j >= 0 && j < SEQ) {
      v4h vv = *((const v4h*)(V + (brow + j) * HDIM) + lane);
#pragma unroll
      for (int i = 0; i < 4; ++i) ctx[i] += wgt[m] * (float)vv[i];
    }
  }
#pragma unroll
  for (int i = 0; i < 4; ++i) lds_ctx[wave][lane * 4 + i] = ctx[i];

  float acc2 = b2[lane];
#pragma unroll 4
  for (int h = 0; h < HDIM; ++h) acc2 += lds_ctx[wave][h] * W2[h * MIDD + lane];
  acc2 = fmaxf(acc2, 0.0f);
  lds_mid[wave][lane] = acc2;

  if (lane < OUTD) {
    float o = b3[lane];
#pragma unroll
    for (int i = 0; i < MIDD; ++i) o += lds_mid[wave][i] * W3[i * OUTD + lane];
    out[((size_t)b * SEQ + s) * OUTD + lane] = o;
  }
}

// ---------------------------------------------------------------------------
extern "C" void kernel_launch(void* const* d_in, const int* in_sizes, int n_in,
                              void* d_out, int out_size, void* d_ws, size_t ws_size,
                              hipStream_t stream) {
  const float* x  = (const float*)d_in[0];
  const float* W1 = (const float*)d_in[1];
  const float* b1 = (const float*)d_in[2];
  const float* Wq = (const float*)d_in[3];
  const float* Wk = (const float*)d_in[4];
  const float* Wv = (const float*)d_in[5];
  const float* bv = (const float*)d_in[6];
  const float* W2 = (const float*)d_in[7];
  const float* b2 = (const float*)d_in[8];
  const float* W3 = (const float*)d_in[9];
  const float* b3 = (const float*)d_in[10];
  const float* pe = (const float*)d_in[11];
  float* out = (float*)d_out;

  _Float16* Xh  = (_Float16*)d_ws;
  _Float16* Qw  = Xh + REGH;
  _Float16* Kw  = Qw + REGH;
  _Float16* Vw  = Kw + REGH;
  _Float16* Wp  = Vw + REGH;            // 4 packed weights, WPACK halves each
  float*    pep = (float*)(Wp + 4 * WPACK);

  dim3 blk(256);
  // prep (tiny, once per launch)
  k_convert_x<<<dim3((unsigned)(REGH / 256)), blk, 0, stream>>>(x, Xh);
  k_pad_pe   <<<dim3(SPAD * HDIM / 256),      blk, 0, stream>>>(pe, pep);
  k_pack_w   <<<dim3(16),                     blk, 0, stream>>>(W1, Wq, Wk, Wv, Wp);
  // main pipeline
  k_hidden_qkv<<<dim3(BATCH * NMTILE), blk, 0, stream>>>(Xh, Wp, b1, pep, Qw, Kw, Vw);
  k_attn      <<<dim3(BATCH * 13),     blk, 0, stream>>>(Qw, Kw, Vw, bv, W2, b2, W3, b3, out);
}
